// Attention_41472204210295
// MI455X (gfx1250) — matrix operand order repaired and checked by reading, not compiled, not run on hardware
//
#include <hip/hip_runtime.h>

typedef __attribute__((ext_vector_type(16))) _Float16 v16h;
typedef __attribute__((ext_vector_type(8)))  _Float16 v8h;
typedef __attribute__((ext_vector_type(8)))  float    v8f;

#define NN 2
#define TT 2048
#define DD 1024
#define HH 16
#define SS 64
#define NEG_INF_F (-1e9f)

static __device__ __forceinline__ v16h cat8(v8h lo, v8h hi) {
  return __builtin_shufflevector(lo, hi, 0,1,2,3,4,5,6,7,8,9,10,11,12,13,14,15);
}

static __device__ __forceinline__ v8f wmma_f16(v16h a, v16h b, v8f c) {
  // D = A(16x32 f16) * B(32x16 f16) + C(16x16 f32)
  return __builtin_amdgcn_wmma_f32_16x16x32_f16(false, a, false, b, (short)0, c,
                                                false, false);
}

// ---------------------------------------------------------------- convert ---
__global__ void convert_f32_to_f16(const float* __restrict__ src,
                                   _Float16* __restrict__ dst,
                                   int n, float scale) {
  int i = blockIdx.x * blockDim.x + threadIdx.x;
  int stride = gridDim.x * blockDim.x;
  for (; i < n; i += stride) dst[i] = (_Float16)(src[i] * scale);
}

// f32 [R,C] row-major -> f16 [C,R] row-major (weight transpose)
__global__ void convert_transpose_f16(const float* __restrict__ src,
                                      _Float16* __restrict__ dst,
                                      int R, int C, float scale) {
  int i = blockIdx.x * blockDim.x + threadIdx.x;
  int n = R * C;
  int stride = gridDim.x * blockDim.x;
  for (; i < n; i += stride) {
    int r = i / C, c = i % C;
    dst[(size_t)c * R + r] = (_Float16)(src[i] * scale);
  }
}

// ---------------------------------------------------------------- GEMM ------
// C[M,1024] = A[M,1024] * B, with B given TRANSPOSED: BT[1024,1024] row-major.
// Each wave computes a 32x32 block (2x2 WMMA tiles): every fetched fragment
// feeds two WMMAs (8 b128 loads -> 4 WMMAs per k-step).
// MODE 0: f16 scattered [N,H,T,S] (Q,K)   MODE 2: f16 scattered [N,H,S,T] (V)
// MODE 1: f32 row-major [M,1024] (final output)
template <int MODE>
__global__ __launch_bounds__(256, 1)
void gemm_wmma(const _Float16* __restrict__ A,
               const _Float16* __restrict__ BT,
               void* __restrict__ dst) {
  constexpr int K = 1024, NC = 1024, M = NN * TT;
  const int lane  = threadIdx.x & 31;
  const int wid   = threadIdx.x >> 5;
  const int mrow  = lane & 15;   // A row / B-C column within tile
  const int kh    = lane >> 4;   // K-half selector
  int sb = blockIdx.x * (blockDim.x >> 5) + wid;   // 32x32 superblock index
  int tm = sb / (NC >> 5), tn = sb % (NC >> 5);
  if (tm * 32 >= M) return;

  v8f c00 = {}, c01 = {}, c10 = {}, c11 = {};
  const _Float16* arow0 = A  + (size_t)(tm * 32 + mrow) * K;
  const _Float16* arow1 = arow0 + (size_t)16 * K;
  const _Float16* brow0 = BT + (size_t)(tn * 32 + mrow) * K;
  const _Float16* brow1 = brow0 + (size_t)16 * K;
#pragma unroll 2
  for (int k0 = 0; k0 < K; k0 += 32) {
    v16h a0 = cat8(*(const v8h*)(arow0 + k0 + kh * 16),
                   *(const v8h*)(arow0 + k0 + kh * 16 + 8));
    v16h a1 = cat8(*(const v8h*)(arow1 + k0 + kh * 16),
                   *(const v8h*)(arow1 + k0 + kh * 16 + 8));
    v16h b0 = cat8(*(const v8h*)(brow0 + k0 + kh * 16),
                   *(const v8h*)(brow0 + k0 + kh * 16 + 8));
    v16h b1 = cat8(*(const v8h*)(brow1 + k0 + kh * 16),
                   *(const v8h*)(brow1 + k0 + kh * 16 + 8));
    c00 = wmma_f16(a0, b0, c00);
    c01 = wmma_f16(a0, b1, c01);
    c10 = wmma_f16(a1, b0, c10);
    c11 = wmma_f16(a1, b1, c11);
  }

  v8f* cs[4] = {&c00, &c01, &c10, &c11};
#pragma unroll
  for (int i = 0; i < 2; ++i)
#pragma unroll
    for (int j = 0; j < 2; ++j) {
      v8f& c = *cs[i * 2 + j];
#pragma unroll
      for (int g = 0; g < 8; ++g) {
        int row = tm * 32 + i * 16 + g + 8 * kh;   // C layout: VGPR g -> rows g, g+8
        int col = tn * 32 + j * 16 + mrow;
        int n = row >> 11, t = row & (TT - 1);     // row = n*T + t
        int h = col >> 6,  s = col & (SS - 1);     // col = h*S + s
        if (MODE == 0) {
          ((_Float16*)dst)[(((size_t)n * HH + h) * TT + t) * SS + s] = (_Float16)c[g];
        } else if (MODE == 2) {
          ((_Float16*)dst)[(((size_t)n * HH + h) * SS + s) * TT + t] = (_Float16)c[g];
        } else {
          ((float*)dst)[(size_t)row * NC + col] = c[g];
        }
      }
    }
}

// ------------------------------------------------------- flash attention ----
// One wave = one (n,h,16-row q-tile); loops r in steps of 32.
// Qh/Kh are [N,H,T,S] f16 (S^-0.5 folded into Q); VhT is [N,H,S,T] f16.
// Oh is [N,T,H*S] f16.  launch_bounds(...,1): allow full VGPR budget, no spills
// (kernel is mask-bandwidth-bound; occupancy is not the limiter).
__global__ __launch_bounds__(256, 1)
void attn_flash(const _Float16* __restrict__ Qh,
                const _Float16* __restrict__ Kh,
                const _Float16* __restrict__ VhT,
                const float*    __restrict__ mask,
                _Float16*       __restrict__ Oh) {
  __shared__ _Float16 plds[8][16 * 32];
  const int lane = threadIdx.x & 31;
  const int wid  = threadIdx.x >> 5;
  const int ncol = lane & 15;
  const int kh   = lane >> 4;

  int tile = blockIdx.x * 8 + wid;
  int nh = tile >> 7;            // (T/16)=128 q-tiles per (n,h)
  int q0 = (tile & 127) << 4;

  const _Float16* Qb  = Qh  + (size_t)nh * TT * SS;
  const _Float16* Kb  = Kh  + (size_t)nh * TT * SS;
  const _Float16* VbT = VhT + (size_t)nh * SS * TT;
  const float*    Mb  = mask + (size_t)nh * TT * TT;

  // Persistent Q A-fragments for K-steps 0..31 and 32..63 (S = 64)
  const _Float16* qrow = Qb + (size_t)(q0 + ncol) * SS;
  v16h aQ0 = cat8(*(const v8h*)(qrow + kh * 16),      *(const v8h*)(qrow + 8 + kh * 16));
  v16h aQ1 = cat8(*(const v8h*)(qrow + 32 + kh * 16), *(const v8h*)(qrow + 40 + kh * 16));

  v8f o[4]; float mi[8], li[8];
#pragma unroll
  for (int t = 0; t < 4; ++t) o[t] = (v8f){};
#pragma unroll
  for (int g = 0; g < 8; ++g) { mi[g] = -3.0e38f; li[g] = 0.f; }

  // mask base for this lane: per-g offsets are compile-time (g*TT, +16)
  const float* mbase = Mb + (size_t)(q0 + 8 * kh) * TT + ncol;

  for (int r0 = 0; r0 < TT; r0 += 32) {
    // ---- scores: two 16x16 tiles (cols r0..+15, r0+16..+31), K-dim = S = 64
    v8f s0 = {}, s1 = {};
    {
      const _Float16* kr = Kb + (size_t)(r0 + ncol) * SS + kh * 16;
      s0 = wmma_f16(aQ0, cat8(*(const v8h*)(kr),      *(const v8h*)(kr + 8)),  s0);
      s0 = wmma_f16(aQ1, cat8(*(const v8h*)(kr + 32), *(const v8h*)(kr + 40)), s0);
    }
    {
      const _Float16* kr = Kb + (size_t)(r0 + 16 + ncol) * SS + kh * 16;
      s1 = wmma_f16(aQ0, cat8(*(const v8h*)(kr),      *(const v8h*)(kr + 8)),  s1);
      s1 = wmma_f16(aQ1, cat8(*(const v8h*)(kr + 32), *(const v8h*)(kr + 40)), s1);
    }

    // prefetch next mask block while we do math (mask is the HBM roof)
    if (r0 + 32 < TT) __builtin_prefetch(mbase + r0 + 32, 0, 1);

    // ---- mask + online softmax (row = g + 8*kh across the 16-lane group)
#pragma unroll
    for (int g = 0; g < 8; ++g) {
      const float* mp = mbase + (size_t)g * TT + r0;
      s0[g] += mp[0]  * NEG_INF_F;
      s1[g] += mp[16] * NEG_INF_F;

      float v = fmaxf(s0[g], s1[g]);
#pragma unroll
      for (int off = 1; off < 16; off <<= 1) v = fmaxf(v, __shfl_xor(v, off, 32));
      float nm    = fmaxf(mi[g], v);
      float alpha = __expf(mi[g] - nm);
      float p0    = __expf(s0[g] - nm);
      float p1    = __expf(s1[g] - nm);
      float rs = p0 + p1;
#pragma unroll
      for (int off = 1; off < 16; off <<= 1) rs += __shfl_xor(rs, off, 32);
      li[g] = li[g] * alpha + rs;
      mi[g] = nm;
      o[0][g] *= alpha; o[1][g] *= alpha; o[2][g] *= alpha; o[3][g] *= alpha;

      int prow = g + 8 * kh;
      plds[wid][prow * 32 + ncol]      = (_Float16)p0;
      plds[wid][prow * 32 + 16 + ncol] = (_Float16)p1;
    }

    // ---- hoist all 4 V B-fragments (contiguous in VhT) so their loads
    //      overlap the barrier + LDS reads; the 4 PV WMMAs then issue together
    v16h bV[4];
#pragma unroll
    for (int t = 0; t < 4; ++t) {
      const _Float16* vp = VbT + (size_t)(t * 16 + ncol) * TT + r0 + kh * 16;
      bV[t] = cat8(*(const v8h*)(vp), *(const v8h*)(vp + 8));
    }

    __syncthreads();

    // ---- refetch P as a WMMA A-fragment (16x32, K = r within block)
    v16h aP;
#pragma unroll
    for (int j = 0; j < 8; ++j) {
      aP[j]     = plds[wid][ncol * 32 + kh * 16 + j];
      aP[8 + j] = plds[wid][ncol * 32 + kh * 16 + 8 + j];
    }
    __syncthreads();

    // ---- O += P(16x32) * V(32x64): four 16x16 tiles
#pragma unroll
    for (int t = 0; t < 4; ++t) o[t] = wmma_f16(aP, bV[t], o[t]);
  }

  // ---- normalize (rcp instead of full divide; f16 output) and store Oh
  float inv[8];
#pragma unroll
  for (int g = 0; g < 8; ++g) inv[g] = __builtin_amdgcn_rcpf(li[g]);
  int n = nh >> 4, h = nh & 15;
#pragma unroll
  for (int t = 0; t < 4; ++t)
#pragma unroll
    for (int g = 0; g < 8; ++g) {
      int rowg = q0 + g + 8 * kh;
      float val = o[t][g] * inv[g];
      Oh[((size_t)n * TT + rowg) * DD + h * SS + t * 16 + ncol] = (_Float16)val;
    }
}

// ---------------------------------------------------------------- launch ----
extern "C" void kernel_launch(void* const* d_in, const int* in_sizes, int n_in,
                              void* d_out, int out_size, void* d_ws, size_t ws_size,
                              hipStream_t stream) {
  (void)in_sizes; (void)n_in; (void)out_size; (void)ws_size;
  const float* q   = (const float*)d_in[0];
  const float* r   = (const float*)d_in[1];
  const float* msk = (const float*)d_in[2];
  const float* Wq  = (const float*)d_in[3];
  const float* Wk  = (const float*)d_in[4];
  const float* Wv  = (const float*)d_in[5];
  const float* Wo  = (const float*)d_in[6];

  const size_t SEQ = (size_t)NN * TT * DD;  // 4M elems
  const size_t WSZ = (size_t)DD * HH * SS;  // 1M elems
  _Float16* p   = (_Float16*)d_ws;
  _Float16* qf  = p; p += SEQ;
  _Float16* rf  = p; p += SEQ;
  _Float16* wqT = p; p += WSZ;   // [HS, D] = B^T for Q projection
  _Float16* wkT = p; p += WSZ;
  _Float16* wvT = p; p += WSZ;
  _Float16* woT = p; p += WSZ;   // [D, HS] = B^T for output projection
  _Float16* Qh  = p; p += SEQ;   // [N,H,T,S]
  _Float16* Kh  = p; p += SEQ;   // [N,H,T,S]
  _Float16* VhT = p; p += SEQ;   // [N,H,S,T]
  _Float16* Oh  = qf;            // qf dead after first GEMM; reuse

  const int CT = 512;
  convert_f32_to_f16<<<(int)((SEQ + CT - 1) / CT), CT, 0, stream>>>(q, qf, (int)SEQ, 1.0f);
  convert_f32_to_f16<<<(int)((SEQ + CT - 1) / CT), CT, 0, stream>>>(r, rf, (int)SEQ, 1.0f);
  // weights: convert + transpose; fold S^-0.5 into Wq
  convert_transpose_f16<<<(int)((WSZ + CT - 1) / CT), CT, 0, stream>>>(Wq, wqT, DD, HH * SS, 0.125f);
  convert_transpose_f16<<<(int)((WSZ + CT - 1) / CT), CT, 0, stream>>>(Wk, wkT, DD, HH * SS, 1.0f);
  convert_transpose_f16<<<(int)((WSZ + CT - 1) / CT), CT, 0, stream>>>(Wv, wvT, DD, HH * SS, 1.0f);
  convert_transpose_f16<<<(int)((WSZ + CT - 1) / CT), CT, 0, stream>>>(Wo, woT, HH * SS, DD, 1.0f);

  const int M = NN * TT;                          // 4096
  const int gblocks = (M / 32) * (DD / 32) / 8;   // 2x2-tile superblocks, 8 waves/block
  gemm_wmma<0><<<gblocks, 256, 0, stream>>>(qf, wqT, (void*)Qh);
  gemm_wmma<0><<<gblocks, 256, 0, stream>>>(rf, wkT, (void*)Kh);
  gemm_wmma<2><<<gblocks, 256, 0, stream>>>(rf, wvT, (void*)VhT);

  const int qtiles = NN * HH * (TT / 16);         // 4096 q-tiles
  attn_flash<<<qtiles / 8, 256, 0, stream>>>(Qh, Kh, VhT, msk, Oh);

  gemm_wmma<1><<<gblocks, 256, 0, stream>>>(Oh, woT, d_out);
}
